// ModuleCorrelationCPU_44547400794796
// MI455X (gfx1250) — matrix operand order repaired and checked by reading, not compiled, not run on hardware
//
#include <hip/hip_runtime.h>

typedef __attribute__((ext_vector_type(16))) __bf16 v16bf;
typedef __attribute__((ext_vector_type(8)))  float  v8f;

#define B_   4
#define C_   64
#define H_   192
#define W_   320
#define TX   16
#define TY   8
#define SROWS (TY + 8)   // 16 rows of second-window (dy in -4..+4 over 8 output rows)
#define SXW   32         // second window width: x0-8 .. x0+23
#define SCP   72         // padded channel pitch (bf16) -> 144B row stride, 16B aligned
#define SDP   20         // D-scratch pixel pitch (f32): 16B aligned, low-conflict
#define SDS   (16 * SDP) // one 16x16 D tile (f32) with pitch

__global__ __launch_bounds__(256)
void corr81_wmma_bf16(const float* __restrict__ first,
                      const float* __restrict__ second,
                      float* __restrict__ out)
{
    __shared__ float  sF[C_ * TY * TX];          // 32 KB f32  [c][y][x], async DMA fill
    __shared__ __bf16 sS[SROWS * SXW * SCP];     // 72 KB bf16 [row][x][c] (pitch 72)
    __shared__ float  sD[8 * 2 * SDS];           // 20 KB f32  per-wave D staging

    const int tid = threadIdx.x;
    const int x0  = blockIdx.x * TX;
    const int y0  = blockIdx.y * TY;
    const int b   = blockIdx.z;

    // ---------- async DMA: first tile (f32, no halo, always in-bounds) ----------
    {
        unsigned ldsF = (unsigned)(unsigned long long)(&sF[0]);
#pragma unroll
        for (int k = 0; k < 8; ++k) {
            int q  = tid + 256 * k;        // 2048 chunks of 4 floats (16B)
            int xq = (q & 3) * 4;
            int t  = q >> 2;
            int y  = t & (TY - 1);
            int c  = t >> 3;
            unsigned long long ga = (unsigned long long)
                (first + (((b * C_ + c) * H_ + (y0 + y)) * W_ + x0 + xq));
            unsigned la = ldsF + 4u * (unsigned)((c * TY + y) * TX + xq);
            asm volatile("global_load_async_to_lds_b128 %0, %1, off"
                         :: "v"(la), "v"(ga) : "memory");
        }
    }

    // ---------- second halo window -> LDS bf16 (vectorized; interior fast path) ----------
    {
        const float* sec = second + (size_t)b * C_ * H_ * W_;
        const bool interior = (x0 >= 8) && (x0 + 24 <= W_) && (y0 >= 4) && (y0 + 12 <= H_);
        if (interior) {
            for (int q = tid; q < SROWS * C_ * 8; q += 256) {
                int xc  = (q & 7) * 4;            // x chunk start within 32-wide window
                int c   = (q >> 3) & (C_ - 1);
                int row = q >> 9;                 // 0..15
                const float4 v = *(const float4*)
                    (sec + ((size_t)c * H_ + (y0 - 4 + row)) * W_ + (x0 - 8) + xc);
                int base = (row * SXW + xc) * SCP + c;
                sS[base          ] = (__bf16)v.x;
                sS[base +     SCP] = (__bf16)v.y;
                sS[base + 2 * SCP] = (__bf16)v.z;
                sS[base + 3 * SCP] = (__bf16)v.w;
            }
        } else {
            for (int q = tid; q < SROWS * C_ * 8; q += 256) {
                int xc  = (q & 7) * 4;
                int c   = (q >> 3) & (C_ - 1);
                int row = q >> 9;
                int gy  = y0 - 4 + row;
                const bool yok = (unsigned)gy < (unsigned)H_;
#pragma unroll
                for (int e = 0; e < 4; ++e) {
                    int gx = x0 - 8 + xc + e;
                    float v = 0.0f;
                    if (yok && (unsigned)gx < (unsigned)W_)
                        v = sec[((size_t)c * H_ + gy) * W_ + gx];
                    sS[(row * SXW + xc + e) * SCP + c] = (__bf16)v;
                }
            }
        }
    }

    asm volatile("s_wait_asynccnt 0" ::: "memory");
    __syncthreads();

    // ---------- per-wave WMMA cross-correlation ----------
    const int lane = tid & 31;
    const int wv   = tid >> 5;      // wave id = output row r (0..7)
    const int hi   = lane >> 4;     // half-wave
    const int nm   = lane & 15;     // M for A-frag, N for B/C/D-frag

    // A fragments: lane(m=nm,hi), elem e: K = e + 16*hi (contiguous k order, same as B)
    v16bf a0, a1;
#pragma unroll
    for (int e = 0; e < 16; ++e) {
        int kb = e + 16 * hi;                              // 0..31
        a0[e] = (__bf16)sF[((kb     ) * TY + wv) * TX + nm];
        a1[e] = (__bf16)sF[((kb + 32) * TY + wv) * TX + nm];
    }

    const float scale = 1.0f / (float)C_;
    const int   yrow  = y0 + wv;
    float* dwave = &sD[wv * 2 * SDS];

    // dy-invariant extraction offsets: pass j reads D element for channel dxi=2j+hi, pixel nm.
    // Laundered through an asm barrier so the compiler keeps them in VGPRs instead of
    // rematerializing the cmp/cndmask chains in every unrolled dy iteration.
    int dOff[5];
#pragma unroll
    for (int j = 0; j < 5; ++j) {
        int dxi = 2 * j + hi;            // 0..9 (9 handled by mask at store)
        int md  = nm + dxi - 4;          // m + delta
        int s   = (md > 11) ? 1 : 0;     // strip select
        int n   = md + 4 - 12 * s;       // column inside selected strip
        dOff[j] = s * SDS + n * SDP + nm;
        asm("" : "+v"(dOff[j]));
    }

    // Loop-carried bases (strength-reduced instead of per-dy v_mad chains).
    const __bf16* r0 = &sS[(wv * SXW + ( 4 + nm)) * SCP + 16 * hi]; // strip x0-4..x0+11
    const __bf16* r1 = r0 + 12 * SCP;                               // strip x0+8..x0+23
    float* d0 = dwave + nm * SDP + 8 * hi;                          // D staging slot
    int obase = ((b * 81) * H_ + yrow) * W_ + x0 + hi * (H_ * W_) + nm;

#pragma unroll
    for (int dy = 0; dy < 9; ++dy) {
        // B fragments: lane n=nm, elem e: K = e + 16*hi (+32 for second half)
        v16bf b00, b01, b10, b11;
#pragma unroll
        for (int e = 0; e < 16; ++e) {
            b00[e] = r0[e];
            b01[e] = r0[e + 32];
            b10[e] = r1[e];
            b11[e] = r1[e + 32];
        }

        v8f acc0 = {0.f,0.f,0.f,0.f,0.f,0.f,0.f,0.f};
        v8f acc1 = {0.f,0.f,0.f,0.f,0.f,0.f,0.f,0.f};
        acc0 = __builtin_amdgcn_wmma_f32_16x16x32_bf16(false, a0, false, b00, (short)0, acc0, false, false);
        acc0 = __builtin_amdgcn_wmma_f32_16x16x32_bf16(false, a1, false, b01, (short)0, acc0, false, false);
        acc1 = __builtin_amdgcn_wmma_f32_16x16x32_bf16(false, a0, false, b10, (short)0, acc1, false, false);
        acc1 = __builtin_amdgcn_wmma_f32_16x16x32_bf16(false, a1, false, b11, (short)0, acc1, false, false);

        // Stage D tiles to per-wave LDS scratch: sD[n][m] (pitch SDP), strips back-to-back.
#pragma unroll
        for (int rr = 0; rr < 8; ++rr) {
            d0[rr]       = acc0[rr];
            d0[SDS + rr] = acc1[rr];
        }

        // Coalesced extraction: pass j writes channels dxi=2j / 2j+1, 16 contiguous pixels each.
#pragma unroll
        for (int j = 0; j < 5; ++j) {
            float v = dwave[dOff[j]] * scale;
            if (j < 4) {
                out[obase + (2 * j) * (H_ * W_)] = v;
            } else if (hi == 0) {          // only dxi=8 is valid in the last pass
                out[obase + 8 * (H_ * W_)] = v;
            }
        }

        r0    += SXW * SCP;       // next source row
        r1    += SXW * SCP;
        obase += 9 * (H_ * W_);   // next dy channel group
    }
}

extern "C" void kernel_launch(void* const* d_in, const int* in_sizes, int n_in,
                              void* d_out, int out_size, void* d_ws, size_t ws_size,
                              hipStream_t stream) {
    const float* first  = (const float*)d_in[0];
    const float* second = (const float*)d_in[1];
    float*       outp   = (float*)d_out;
    dim3 grid(W_ / TX, H_ / TY, B_);   // (20, 24, 4)
    corr81_wmma_bf16<<<grid, dim3(256), 0, stream>>>(first, second, outp);
}
